// RelativeGridAttention_39943195853190
// MI455X (gfx1250) — matrix operand order repaired and checked by reading, not compiled, not run on hardware
//
#include <hip/hip_runtime.h>

typedef __attribute__((ext_vector_type(16))) __bf16 bf16x16;
typedef __attribute__((ext_vector_type(8)))  __bf16 bf16x8;
typedef __attribute__((ext_vector_type(8)))  float  f32x8;

namespace {
constexpr int Bn = 2, Hn = 8, Qn = 2048, Kn = 4096, Dn = 128, G = 64;
constexpr int TBL  = 2 * G - 1;      // 127
constexpr int SROW = Kn + 4;         // padded LDS row stride (floats)
constexpr float SCALE = 0.08838834764831845f;  // 1/sqrt(128)
constexpr size_t QELEMS = (size_t)Bn * Hn * Qn * Dn;   // 4,194,304
constexpr size_t KELEMS = (size_t)Bn * Hn * Kn * Dn;   // 8,388,608
}

// ---------- pass 0: f32 -> bf16 conversion (8 elements / thread) ----------
__global__ __launch_bounds__(256)
void cvt_bf16_kernel(const float* __restrict__ src, __bf16* __restrict__ dst, int n8) {
  const int i = blockIdx.x * 256 + threadIdx.x;
  if (i >= n8) return;
  const float4* p = (const float4*)src + 2 * (size_t)i;
  const float4 a = p[0];
  const float4 b = p[1];
  bf16x8 v;
  v[0] = (__bf16)a.x; v[1] = (__bf16)a.y; v[2] = (__bf16)a.z; v[3] = (__bf16)a.w;
  v[4] = (__bf16)b.x; v[5] = (__bf16)b.y; v[6] = (__bf16)b.z; v[7] = (__bf16)b.w;
  *((bf16x8*)dst + i) = v;
}

// ---------- pass 1: scores (WMMA) + bias + fused softmax ----------
__global__ __launch_bounds__(256)
void rga_wmma_kernel(const __bf16* __restrict__ qbf,
                     const __bf16* __restrict__ kbf,
                     const int*    __restrict__ pos,
                     const float*  __restrict__ rel_bias,
                     float*        __restrict__ out)
{
  extern __shared__ float s_scores[];          // [16][SROW] ~256 KB of the 320 KB WGP LDS
  __shared__ int s_posx[16];
  __shared__ int s_posy[16];

  const int blk = blockIdx.x;                  // 2048 blocks: (b, h, qtile)
  const int qt  = blk & (Qn / 16 - 1);
  const int h   = (blk >> 7) & (Hn - 1);
  const int b   = blk >> 10;
  const int q0  = qt * 16;

  const int tid  = threadIdx.x;
  const int wave = tid >> 5;                   // 8 wave32s
  const int lane = tid & 31;
  const int m16  = lane & 15;
  const int hi   = lane >> 4;

  if (tid < 16) {
    s_posx[tid] = pos[(b * Qn + q0 + tid) * 2 + 0];
    s_posy[tid] = pos[(b * Qn + q0 + tid) * 2 + 1];
  }

  // --- A operand: 16x128 Q tile as 4 chunks of 16x32 bf16, contiguous k = 16*hi + s (same order in B) ---
  // lane L (L<16): row M=L, K = {0..15}; lane L+16: row M=L, K = {16..31}
  const __bf16* qrow = qbf + ((size_t)(b * Hn + h) * Qn + q0 + m16) * Dn;
  bf16x16 A[4];
#pragma unroll
  for (int c = 0; c < 4; ++c) {
    const int d0 = c * 32 + 16 * hi;
    const bf16x8 lo = *(const bf16x8*)(qrow + d0);
    const bf16x8 hi8 = *(const bf16x8*)(qrow + d0 + 8);
    A[c] = __builtin_shufflevector(lo, hi8, 0, 1, 2, 3, 4, 5, 6, 7,
                                            8, 9, 10, 11, 12, 13, 14, 15);
  }

  const __bf16* kbase = kbf + (size_t)(b * Hn + h) * Kn * Dn;

  // Each wave owns 512 key columns = 32 column tiles of 16.
  for (int t = 0; t < 32; ++t) {
    const int kcol = wave * 512 + t * 16 + m16;
    // B operand 32x16 bf16: lane L holds col N=L%16, K = 16*hi + {0..15} (+32*chunk)
    const __bf16* krow = kbase + (size_t)kcol * Dn;
    if (t < 31) __builtin_prefetch(krow + 16 * Dn, 0, 3);  // global_prefetch next tile

    f32x8 acc = {};
#pragma unroll
    for (int c = 0; c < 4; ++c) {
      const bf16x16 Bv = *(const bf16x16*)(krow + c * 32 + 16 * hi);
      acc = __builtin_amdgcn_wmma_f32_16x16x32_bf16(
          false, A[c], false, Bv, (short)0, acc, false, false);
    }

    // C/D layout: element r -> row M = r + 8*hi, col N = lane%16 (raw scores, bias later)
#pragma unroll
    for (int r = 0; r < 8; ++r)
      s_scores[(r + 8 * hi) * SROW + kcol] = acc[r];
  }
  __syncthreads();

  // --- bias + fused softmax over K=4096, 2 rows per wave ---
  const float* btab = rel_bias + (size_t)h * TBL * TBL;
  for (int row = wave; row < 16; row += 8) {
    float* srow = s_scores + row * SROW;
    const int px = s_posx[row];
    const int py = s_posy[row];
    // bias element for k: btab[(k/64 - py + 63)*127 + (k%64 - px + 63)]; indices provably
    // in [0,126] for this problem, so the reference's clip is a no-op.
    const float* bb = btab + (size_t)(G - 1 - py) * TBL + (G - 1 - px) + lane;

    float mx = -3.4e38f;
    for (int gy = 0; gy < G; ++gy) {
      const float* brow = bb + gy * TBL;       // coalesced: consecutive lanes -> consecutive addrs
      const int i0 = gy * 64 + lane;
      const float s0 = srow[i0]      * SCALE + brow[0];
      const float s1 = srow[i0 + 32] * SCALE + brow[32];
      srow[i0]      = s0;
      srow[i0 + 32] = s1;
      mx = fmaxf(mx, fmaxf(s0, s1));
    }
#pragma unroll
    for (int o = 16; o > 0; o >>= 1) mx = fmaxf(mx, __shfl_xor(mx, o, 32));

    float sum = 0.f;
    for (int i = lane; i < Kn; i += 32) {
      const float e = __expf(srow[i] - mx);
      srow[i] = e;
      sum += e;
    }
#pragma unroll
    for (int o = 16; o > 0; o >>= 1) sum += __shfl_xor(sum, o, 32);

    const float inv = 1.0f / sum;
    float* orow = out + ((size_t)(b * Hn + h) * Qn + q0 + row) * (size_t)Kn;
    for (int i = lane; i < Kn; i += 32) orow[i] = srow[i] * inv;
  }
}

extern "C" void kernel_launch(void* const* d_in, const int* in_sizes, int n_in,
                              void* d_out, int out_size, void* d_ws, size_t ws_size,
                              hipStream_t stream) {
  (void)in_sizes; (void)n_in; (void)ws_size; (void)out_size;
  const float* qry = (const float*)d_in[0];
  const float* key = (const float*)d_in[1];
  const int*   pos = (const int*)d_in[2];   // jax default x64-off: int32
  const float* rb  = (const float*)d_in[3];
  // d_in[4], d_in[5] (grid_x/grid_y) recomputed arithmetically on device
  float* out = (float*)d_out;

  // bf16 staging in workspace: 8 MB (Q) + 16 MB (K) = 24 MB
  __bf16* qbf = (__bf16*)d_ws;
  __bf16* kbf = qbf + QELEMS;

  const int q8 = (int)(QELEMS / 8);   // 524,288 threads
  const int k8 = (int)(KELEMS / 8);   // 1,048,576 threads
  cvt_bf16_kernel<<<dim3((q8 + 255) / 256), 256, 0, stream>>>(qry, qbf, q8);
  cvt_bf16_kernel<<<dim3((k8 + 255) / 256), 256, 0, stream>>>(key, kbf, k8);

  const size_t lds = (size_t)16 * SROW * sizeof(float);  // ~256.25 KB dynamic LDS
  (void)hipFuncSetAttribute((const void*)rga_wmma_kernel,
                            hipFuncAttributeMaxDynamicSharedMemorySize, (int)lds);

  dim3 grid(Bn * Hn * (Qn / 16));   // 2048 workgroups
  rga_wmma_kernel<<<grid, 256, lds, stream>>>(qbf, kbf, pos, rb, out);
}
